// MotionEncoder_24592982737587
// MI455X (gfx1250) — compile-verified, matrix operand order checked by reading, not run on hardware
//
#include <hip/hip_runtime.h>
#include <hip/hip_bf16.h>

typedef __attribute__((ext_vector_type(16))) _Float16 v16h;
typedef __attribute__((ext_vector_type(8)))  float    v8f;

#define HID   64
#define HEADS 4
#define CH    16
#define GAT_EPS 1e-5f

union Frag16 { v16h v; _Float16 h[16]; unsigned u[8]; };

// K index pattern for 16-bit WMMA A/B fragments (pair p -> first K of the pair),
// per CDNA5 ISA 7.12.2 (lanes 0-15; lanes 16-31 add +8).
__device__ __forceinline__ int kpat(int p) { return (p < 4) ? 2 * p : 8 + 2 * p; }

__device__ __forceinline__ float sigf(float x)  { return 1.f / (1.f + __expf(-x)); }
__device__ __forceinline__ float tanhf_f(float x){ return 1.f - 2.f / (__expf(2.f * x) + 1.f); }

// order-preserving float <-> uint key (for atomicMax-based segment_max)
__device__ __forceinline__ unsigned fkey(float f) {
    unsigned u = __float_as_uint(f);
    return (u & 0x80000000u) ? ~u : (u | 0x80000000u);
}
__device__ __forceinline__ float fdec(unsigned k) {
    unsigned u = (k & 0x80000000u) ? (k & 0x7fffffffu) : ~k;
    return __uint_as_float(u);
}

__device__ __forceinline__ v8f wmma_f16(const v16h& a, const v16h& b, v8f c) {
    return __builtin_amdgcn_wmma_f32_16x16x32_f16(false, a, false, b, (short)0, c, false, false);
}

// ---------------------------------------------------------------------------
// LSTM: one wave (32 lanes) per tile of 16 batch rows. D = input feature dim.
// MODE 0: output last h. MODE 1: output mean of h over time.
// Gate layout: n in [0,256) = gate*64 + j, gates i,f,g,o.
// ---------------------------------------------------------------------------
template <int D, int MODE>
__global__ void lstm_kernel(const float* __restrict__ x,
                            const float* __restrict__ w_ih,   // [256][D]
                            const float* __restrict__ w_hh,   // [256][64]
                            const float* __restrict__ bias,   // [256]
                            int T, int rows,
                            float* __restrict__ emb_out,      // [rows][64]
                            float* __restrict__ feat_out)     // [rows][64]
{
    __shared__ _Float16 h_lds[16 * 64];          // h tile, [m][k], f16
    __shared__ _Float16 bfrag[16 * 2 * 32 * 16]; // w_hh B-fragments, per (nt,ks,lane)
    __shared__ float    g_lds[256 * 16];         // gate pre-activations [n][m]
    __shared__ float    wih_lds[256 * D];
    __shared__ float    b_lds[256];

    const int lane  = threadIdx.x & 31;
    const int nl    = lane & 15;
    const int half  = lane >> 4;
    const int tile0 = blockIdx.x * 16;

    for (int i = lane; i < 256 * D; i += 32) wih_lds[i] = w_ih[i];
    for (int i = lane; i < 256;     i += 32) b_lds[i]   = bias[i];
    // pre-gather w_hh^T into B-fragment lane order: B[k][n] = w_hh[n*64+k]
    for (int nt = 0; nt < 16; ++nt) {
        int n = nt * 16 + nl;
        for (int ks = 0; ks < 2; ++ks) {
            _Float16* dst = &bfrag[((nt * 2 + ks) * 32 + lane) * 16];
#pragma unroll
            for (int p = 0; p < 8; ++p) {
                int k = kpat(p) + 8 * half + 32 * ks;
                dst[2 * p]     = (_Float16)w_hh[n * 64 + k];
                dst[2 * p + 1] = (_Float16)w_hh[n * 64 + k + 1];
            }
        }
    }
    for (int i = lane; i < 16 * 64; i += 32) h_lds[i] = (_Float16)0.f;
    __syncthreads();

    float c[4][8], s[4][8];
#pragma unroll
    for (int jt = 0; jt < 4; ++jt)
#pragma unroll
        for (int r = 0; r < 8; ++r) { c[jt][r] = 0.f; s[jt][r] = 0.f; }

    for (int t = 0; t < T; ++t) {
        // per-thread x values for the 8 rows it owns
        float xv[8][D];
#pragma unroll
        for (int r = 0; r < 8; ++r) {
            int row = tile0 + r + 8 * half;
            if (row >= rows) row = rows - 1;
#pragma unroll
            for (int d = 0; d < D; ++d)
                xv[r][d] = x[(size_t)row * T * D + t * D + d];
        }
        // A fragments from h_lds (m = nl, K offset +8 for high half-wave)
        Frag16 a0, a1;
#pragma unroll
        for (int p = 0; p < 8; ++p) {
            int k = kpat(p) + 8 * half;
            a0.u[p] = *(const unsigned*)&h_lds[nl * 64 + k];
            a1.u[p] = *(const unsigned*)&h_lds[nl * 64 + k + 32];
        }
        // gate GEMM: 16 N-tiles x 2 K-steps of v_wmma_f32_16x16x32_f16
        for (int nt = 0; nt < 16; ++nt) {
            int n = nt * 16 + nl;
            float bb = b_lds[n];
            float wv[D];
#pragma unroll
            for (int d = 0; d < D; ++d) wv[d] = wih_lds[n * D + d];
            v8f acc;
#pragma unroll
            for (int r = 0; r < 8; ++r) {
                float sum = bb;
#pragma unroll
                for (int d = 0; d < D; ++d) sum += xv[r][d] * wv[d];
                acc[r] = sum;
            }
            Frag16 b0, b1;
            b0.v = *(const v16h*)&bfrag[((nt * 2 + 0) * 32 + lane) * 16];
            b1.v = *(const v16h*)&bfrag[((nt * 2 + 1) * 32 + lane) * 16];
            acc = wmma_f16(a0.v, b0.v, acc);
            acc = wmma_f16(a1.v, b1.v, acc);
#pragma unroll
            for (int r = 0; r < 8; ++r)
                g_lds[n * 16 + r + 8 * half] = acc[r];
        }
        __syncthreads();
        // gate nonlinearity + state update (all 4 gates of a cell in same lane)
        float inv = 1.f;
#pragma unroll
        for (int jt = 0; jt < 4; ++jt) {
            int j = jt * 16 + nl;
#pragma unroll
            for (int r = 0; r < 8; ++r) {
                int m = r + 8 * half;
                float gi = g_lds[(j      ) * 16 + m];
                float gf = g_lds[(j +  64) * 16 + m];
                float gg = g_lds[(j + 128) * 16 + m];
                float go = g_lds[(j + 192) * 16 + m];
                float cc = sigf(gf) * c[jt][r] + sigf(gi) * tanhf_f(gg);
                c[jt][r] = cc;
                float hh = sigf(go) * tanhf_f(cc);
                if (MODE == 1) s[jt][r] += hh; else s[jt][r] = hh;
                h_lds[m * 64 + j] = (_Float16)hh;
            }
        }
        __syncthreads();
        (void)inv;
    }

    const float scale = (MODE == 1) ? (1.f / (float)T) : 1.f;
#pragma unroll
    for (int jt = 0; jt < 4; ++jt) {
        int j = jt * 16 + nl;
#pragma unroll
        for (int r = 0; r < 8; ++r) {
            int row = tile0 + r + 8 * half;
            if (row < rows) {
                float val = s[jt][r] * scale;
                emb_out[(size_t)row * 64 + j]  = val;
                feat_out[(size_t)row * 64 + j] = val;
            }
        }
    }
}

// ---------------------------------------------------------------------------
// GAT dense part: xh = x@W ; res = x@rW + bias. One wave per 16-row tile.
// split==0: residual -> resA[row]; split==1: row<A -> resA, else resL.
// ---------------------------------------------------------------------------
__global__ void gat_gemm_kernel(const float* __restrict__ xfeat,
                                const float* __restrict__ W,   // [64][64] (k,n)
                                const float* __restrict__ rW,  // [64][64]
                                const float* __restrict__ bias,// [64]
                                float* __restrict__ xh,
                                float* __restrict__ resA,
                                float* __restrict__ resL,
                                int N, int A, int split)
{
    int wave  = blockIdx.x * (blockDim.x >> 5) + (threadIdx.x >> 5);
    int lane  = threadIdx.x & 31;
    int tile0 = wave * 16;
    if (tile0 >= N) return;
    int nl = lane & 15, half = lane >> 4;

    int mrow = tile0 + nl;
    if (mrow >= N) mrow = N - 1;
    Frag16 a0, a1;
#pragma unroll
    for (int p = 0; p < 8; ++p) {
        int k = kpat(p) + 8 * half;
        a0.h[2 * p]     = (_Float16)xfeat[(size_t)mrow * 64 + k];
        a0.h[2 * p + 1] = (_Float16)xfeat[(size_t)mrow * 64 + k + 1];
        a1.h[2 * p]     = (_Float16)xfeat[(size_t)mrow * 64 + k + 32];
        a1.h[2 * p + 1] = (_Float16)xfeat[(size_t)mrow * 64 + k + 33];
    }

    for (int nt = 0; nt < 4; ++nt) {
        int n = nt * 16 + nl;
        v8f accW, accR;
        float bb = bias[n];
#pragma unroll
        for (int r = 0; r < 8; ++r) { accW[r] = 0.f; accR[r] = bb; }
#pragma unroll
        for (int ks = 0; ks < 2; ++ks) {
            Frag16 bW, bR;
            int koff = 8 * half + 32 * ks;
#pragma unroll
            for (int p = 0; p < 8; ++p) {
                int k = kpat(p) + koff;
                bW.h[2 * p]     = (_Float16)W[k * 64 + n];
                bW.h[2 * p + 1] = (_Float16)W[(k + 1) * 64 + n];
                bR.h[2 * p]     = (_Float16)rW[k * 64 + n];
                bR.h[2 * p + 1] = (_Float16)rW[(k + 1) * 64 + n];
            }
            const v16h& av = ks ? a1.v : a0.v;
            accW = wmma_f16(av, bW.v, accW);
            accR = wmma_f16(av, bR.v, accR);
        }
#pragma unroll
        for (int r = 0; r < 8; ++r) {
            int row = tile0 + r + 8 * half;
            if (row < N) {
                xh[(size_t)row * 64 + n] = accW[r];
                float* res = (!split) ? (resA + (size_t)row * 64)
                           : (row < A ? resA + (size_t)row * 64
                                      : resL + (size_t)(row - A) * 64);
                res[n] = accR[r];
            }
        }
    }
}

// per (node,head): a_src / a_dst attention coefficients
__global__ void attn_coef_kernel(const float* __restrict__ xh,
                                 const float* __restrict__ attS, // [4][16]
                                 const float* __restrict__ attD,
                                 float* __restrict__ asrc, float* __restrict__ adst, int N)
{
    int gid = blockIdx.x * blockDim.x + threadIdx.x;
    if (gid >= N * HEADS) return;
    int node = gid >> 2, h = gid & 3;
    float ss = 0.f, sd = 0.f;
#pragma unroll
    for (int c = 0; c < CH; ++c) {
        float v = xh[(size_t)node * 64 + h * CH + c];
        ss += v * attS[h * CH + c];
        sd += v * attD[h * CH + c];
    }
    asrc[gid] = ss;
    adst[gid] = sd;
}

__global__ void softmax_init_kernel(unsigned* __restrict__ emaxk, float* __restrict__ denom, int n)
{
    int gid = blockIdx.x * blockDim.x + threadIdx.x;
    if (gid < n) { emaxk[gid] = 0u; denom[gid] = 0.f; }
}

// flat edge id -> (src,dst) over [aa edges | a->l edges | l->a edges]
__device__ __forceinline__ void edge_sd(int e, const int* __restrict__ aa,
                                        const int* __restrict__ al,
                                        int Eaa, int Eal, int A, int& s, int& d)
{
    if (e < Eaa)            { s = aa[e];            d = aa[Eaa + e]; }
    else if ((e -= Eaa) < Eal) { s = al[e];         d = al[Eal + e] + A; }
    else                    { e -= Eal; s = al[Eal + e] + A; d = al[e]; }
}

__device__ __forceinline__ float edge_e(int s, int d, int h,
                                        const float* __restrict__ asrc,
                                        const float* __restrict__ adst)
{
    float v = asrc[s * HEADS + h] + adst[d * HEADS + h];
    return v > 0.f ? v : 0.2f * v; // leaky_relu(0.2)
}

__global__ void edge_max_kernel(const int* __restrict__ aa, const int* __restrict__ al,
                                int Eaa, int Eal, int A, int Etot,
                                const float* __restrict__ asrc, const float* __restrict__ adst,
                                unsigned* __restrict__ emaxk)
{
    int e = blockIdx.x * blockDim.x + threadIdx.x;
    if (e >= Etot) return;
    int s, d; edge_sd(e, aa, al, Eaa, Eal, A, s, d);
#pragma unroll
    for (int h = 0; h < HEADS; ++h)
        atomicMax(&emaxk[d * HEADS + h], fkey(edge_e(s, d, h, asrc, adst)));
}

__global__ void edge_sum_kernel(const int* __restrict__ aa, const int* __restrict__ al,
                                int Eaa, int Eal, int A, int Etot,
                                const float* __restrict__ asrc, const float* __restrict__ adst,
                                const unsigned* __restrict__ emaxk, float* __restrict__ denom)
{
    int e = blockIdx.x * blockDim.x + threadIdx.x;
    if (e >= Etot) return;
    int s, d; edge_sd(e, aa, al, Eaa, Eal, A, s, d);
#pragma unroll
    for (int h = 0; h < HEADS; ++h) {
        float ee = __expf(edge_e(s, d, h, asrc, adst) - fdec(emaxk[d * HEADS + h]));
        atomicAdd(&denom[d * HEADS + h], ee);
    }
}

__global__ void edge_scatter_kernel(const int* __restrict__ aa, const int* __restrict__ al,
                                    int Eaa, int Eal, int A, int Etot,
                                    const float* __restrict__ asrc, const float* __restrict__ adst,
                                    const unsigned* __restrict__ emaxk, const float* __restrict__ denom,
                                    const float* __restrict__ xh,
                                    float* __restrict__ outA, float* __restrict__ outL, int split)
{
    int gid = blockIdx.x * blockDim.x + threadIdx.x;
    if (gid >= Etot * HEADS) return;
    int e = gid >> 2, h = gid & 3;
    int s, d; edge_sd(e, aa, al, Eaa, Eal, A, s, d);
    float ee = __expf(edge_e(s, d, h, asrc, adst) - fdec(emaxk[d * HEADS + h]));
    float alpha = ee / (denom[d * HEADS + h] + 1e-16f);
    const float* xs = xh + (size_t)s * 64 + h * CH;
    float* od = (!split) ? (outA + (size_t)d * 64)
              : (d < A ? outA + (size_t)d * 64 : outL + (size_t)(d - A) * 64);
    od += h * CH;
#pragma unroll
    for (int c = 0; c < CH; ++c)
        atomicAdd(&od[c], alpha * xs[c]);
}

// BatchNorm: one block per column (64 blocks), biased variance
__global__ void bn_reduce_kernel(const float* __restrict__ hbuf, float* __restrict__ stat, int N)
{
    __shared__ float ssum[256], ssq[256];
    int j = blockIdx.x, tid = threadIdx.x;
    float s = 0.f, q = 0.f;
    for (int r = tid; r < N; r += 256) {
        float v = hbuf[(size_t)r * 64 + j];
        s += v; q += v * v;
    }
    ssum[tid] = s; ssq[tid] = q;
    __syncthreads();
    for (int off = 128; off > 0; off >>= 1) {
        if (tid < off) { ssum[tid] += ssum[tid + off]; ssq[tid] += ssq[tid + off]; }
        __syncthreads();
    }
    if (tid == 0) {
        float mu  = ssum[0] / (float)N;
        float var = ssq[0] / (float)N - mu * mu;
        stat[j]      = mu;
        stat[64 + j] = rsqrtf(var + GAT_EPS);
    }
}

__global__ void bn_elu_kernel(const float* __restrict__ hbuf, const float* __restrict__ stat,
                              const float* __restrict__ g, const float* __restrict__ be,
                              float* __restrict__ xfeat, int N)
{
    int gid = blockIdx.x * blockDim.x + threadIdx.x;
    if (gid >= N * 64) return;
    int c = gid & 63;
    float v = (hbuf[gid] - stat[c]) * stat[64 + c] * g[c] + be[c];
    xfeat[gid] = v > 0.f ? v : (__expf(v) - 1.f);
}

// ---------------------------------------------------------------------------
extern "C" void kernel_launch(void* const* d_in, const int* in_sizes, int n_in,
                              void* d_out, int out_size, void* d_ws, size_t ws_size,
                              hipStream_t stream)
{
    const float* agent_hist = (const float*)d_in[0];
    const float* lane_nodes = (const float*)d_in[1];
    const int*   e_aa       = (const int*)d_in[2];
    const int*   e_al       = (const int*)d_in[3];
    const float* aw_ih = (const float*)d_in[4];
    const float* aw_hh = (const float*)d_in[5];
    const float* ab    = (const float*)d_in[6];
    const float* lw_ih = (const float*)d_in[7];
    const float* lw_hh = (const float*)d_in[8];
    const float* lb    = (const float*)d_in[9];
    const float* w1    = (const float*)d_in[10];
    const float* as1   = (const float*)d_in[11];
    const float* ad1   = (const float*)d_in[12];
    const float* rw1   = (const float*)d_in[13];
    const float* b1    = (const float*)d_in[14];
    const float* g1    = (const float*)d_in[15];
    const float* be1   = (const float*)d_in[16];
    const float* w2    = (const float*)d_in[17];
    const float* as2   = (const float*)d_in[18];
    const float* ad2   = (const float*)d_in[19];
    const float* rw2   = (const float*)d_in[20];
    const float* b2    = (const float*)d_in[21];

    const int T_A = 50, T_L = 20;
    const int A    = in_sizes[0] / (T_A * 5);
    const int L    = in_sizes[1] / (T_L * 2);
    const int Eaa  = in_sizes[2] / 2;
    const int Eal  = in_sizes[3] / 2;
    const int N    = A + L;
    const int Etot = Eaa + 2 * Eal;

    float* out = (float*)d_out;
    float* out_agent_emb = out;
    float* out_xa        = out + (size_t)A * 64;
    float* out_lane_emb  = out + (size_t)2 * A * 64;
    float* out_xl        = out + (size_t)2 * A * 64 + (size_t)L * 64;

    // workspace carve-up
    char* wp = (char*)d_ws;
    auto carve = [&](size_t bytes) { void* p = wp; wp += (bytes + 255) & ~(size_t)255; return p; };
    float*    x_feat = (float*)carve((size_t)N * 64 * 4);
    float*    xh     = (float*)carve((size_t)N * 64 * 4);
    float*    hbuf   = (float*)carve((size_t)N * 64 * 4);
    float*    asrc   = (float*)carve((size_t)N * HEADS * 4);
    float*    adst   = (float*)carve((size_t)N * HEADS * 4);
    unsigned* emaxk  = (unsigned*)carve((size_t)N * HEADS * 4);
    float*    denom  = (float*)carve((size_t)N * HEADS * 4);
    float*    bnstat = (float*)carve(128 * 4);

    // ---- LSTM encoders (WMMA over recurrent GEMM) ----
    lstm_kernel<5, 0><<<(A + 15) / 16, 32, 0, stream>>>(
        agent_hist, aw_ih, aw_hh, ab, T_A, A, out_agent_emb, x_feat);
    lstm_kernel<2, 1><<<(L + 15) / 16, 32, 0, stream>>>(
        lane_nodes, lw_ih, lw_hh, lb, T_L, L, out_lane_emb, x_feat + (size_t)A * 64);

    const int ntiles = (N + 15) / 16;
    const int gemm_blocks = (ntiles + 3) / 4;

    auto run_gat = [&](const float* W, const float* attS, const float* attD,
                       const float* rW, const float* bias,
                       float* resA, float* resL, int split) {
        gat_gemm_kernel<<<gemm_blocks, 128, 0, stream>>>(
            x_feat, W, rW, bias, xh, resA, resL, N, A, split);
        attn_coef_kernel<<<(N * HEADS + 255) / 256, 256, 0, stream>>>(
            xh, attS, attD, asrc, adst, N);
        softmax_init_kernel<<<(N * HEADS + 255) / 256, 256, 0, stream>>>(
            emaxk, denom, N * HEADS);
        edge_max_kernel<<<(Etot + 255) / 256, 256, 0, stream>>>(
            e_aa, e_al, Eaa, Eal, A, Etot, asrc, adst, emaxk);
        edge_sum_kernel<<<(Etot + 255) / 256, 256, 0, stream>>>(
            e_aa, e_al, Eaa, Eal, A, Etot, asrc, adst, emaxk, denom);
        edge_scatter_kernel<<<(Etot * HEADS + 255) / 256, 256, 0, stream>>>(
            e_aa, e_al, Eaa, Eal, A, Etot, asrc, adst, emaxk, denom, xh,
            resA, resL, split);
    };

    // ---- GAT layer 1 -> BatchNorm -> ELU ----
    run_gat(w1, as1, ad1, rw1, b1, hbuf, nullptr, 0);
    bn_reduce_kernel<<<64, 256, 0, stream>>>(hbuf, bnstat, N);
    bn_elu_kernel<<<(N * 64 + 255) / 256, 256, 0, stream>>>(hbuf, bnstat, g1, be1, x_feat, N);

    // ---- GAT layer 2 (writes directly into d_out regions) ----
    run_gat(w2, as2, ad2, rw2, b2, out_xa, out_xl, 1);
}
